// MLA_vLLM_24008867184768
// MI455X (gfx1250) — compile-verified, matrix operand order checked by reading, not run on hardware
//
#include <hip/hip_runtime.h>
#include <hip/hip_bf16.h>

// ---------------- types / helpers -------------------------------------------
typedef __bf16 bf16;
typedef __attribute__((ext_vector_type(16))) __bf16 v16bf;
typedef __attribute__((ext_vector_type(8)))  float  v8f;
typedef __attribute__((ext_vector_type(4)))  unsigned v4u;
typedef __attribute__((ext_vector_type(8)))  int      v8i;
typedef __attribute__((ext_vector_type(4)))  int      v4i;

#define D_MODEL 2048
#define H       16
#define Q_LORA  1536
#define KV_LORA 512
#define DN      128
#define DR      64
#define DV      128
#define BSZ     2
#define SEQ     2048
#define ROWS    (BSZ*SEQ)      // 4096
#define CD      (KV_LORA+DR)   // 576

struct alignas(16) BF8 { bf16  v[8]; };
struct alignas(8)  BF4 { bf16  v[4]; };
struct alignas(16) F4  { float v[4]; };
struct alignas(16) U4  { unsigned v[4]; };

#if defined(__gfx1250__) && __has_builtin(__builtin_amdgcn_tensor_load_to_lds)
#define HAVE_TDM 1
#endif

__device__ __forceinline__ v8f wmma_bf16(v16bf a, v16bf b, v8f c) {
  return __builtin_amdgcn_wmma_f32_16x16x32_bf16(false, a, false, b, (short)0, c,
                                                 false, false);
}

// 16-bit operand fragment (A or B) for 16x16x32: lane (idx15, h) holds
// k = 8h..8h+7 and 16+8h..16+8h+7  -> two contiguous 16B chunks.
__device__ __forceinline__ v16bf frag_from_bf16(const bf16* rp, int h) {
  BF8 a = *(const BF8*)(rp + 8 * h);
  BF8 b = *(const BF8*)(rp + 16 + 8 * h);
  v16bf f;
#pragma unroll
  for (int i = 0; i < 8; ++i) { f[i] = a.v[i]; f[8 + i] = b.v[i]; }
  return f;
}

// Software-pipelined GEMM core: NT output tiles per wave, K-chunks of 32.
// Double-buffered B fragments across K iterations so WMMAs never wait on a
// load issued in the same chunk. kchunks must be even.
template <int NT>
__device__ __forceinline__ void gemm_panel(const bf16* arow,
                                           const bf16* const* bbase,
                                           int kchunks, int hh, v8f* acc) {
  v16bf bA[NT], bB[NT];
#pragma unroll
  for (int t = 0; t < NT; ++t) bA[t] = frag_from_bf16(bbase[t], hh);
  for (int kc = 0; kc < kchunks; kc += 2) {
    v16bf a0 = frag_from_bf16(arow + kc * 32, hh);
#pragma unroll
    for (int t = 0; t < NT; ++t)
      bB[t] = frag_from_bf16(bbase[t] + (kc + 1) * 32, hh);
#pragma unroll
    for (int t = 0; t < NT; ++t) acc[t] = wmma_bf16(a0, bA[t], acc[t]);
    v16bf a1 = frag_from_bf16(arow + (kc + 1) * 32, hh);
    if (kc + 2 < kchunks) {
#pragma unroll
      for (int t = 0; t < NT; ++t)
        bA[t] = frag_from_bf16(bbase[t] + (kc + 2) * 32, hh);
    }
#pragma unroll
    for (int t = 0; t < NT; ++t) acc[t] = wmma_bf16(a1, bB[t], acc[t]);
  }
}

// ---- Tensor Data Mover: 2D bf16 tile global->LDS (D# per ISA 08 section 8) --
__device__ __forceinline__ void tdm_load_2d(void* dstLds, const void* src,
                                            unsigned rowElems, unsigned rows,
                                            unsigned strideElems) {
#ifdef HAVE_TDM
  unsigned long long ga = (unsigned long long)src;
  unsigned lds_off = (unsigned)(unsigned long long)dstLds;  // low 32b = LDS byte offset
  v4u g0;
  g0[0] = 1u;                                               // count=1 valid user D#
  g0[1] = lds_off;                                          // lds_addr (bytes)
  g0[2] = (unsigned)(ga & 0xffffffffu);                     // global_addr[31:0]
  g0[3] = (unsigned)((ga >> 32) & 0x01ffffffu) | (2u << 30);// addr[56:32] | type=2
  v8i g1;
  g1[0] = (int)(1u << 16);                                  // data_size=1 (2 bytes)
  g1[1] = (int)((rowElems & 0xffffu) << 16);                // tensor_dim0[15:0]
  g1[2] = (int)((rowElems >> 16) | ((rows & 0xffffu) << 16));// d0 hi | tensor_dim1 lo
  g1[3] = (int)((rows >> 16) | ((rowElems & 0xffffu) << 16));// d1 hi | tile_dim0
  g1[4] = (int)(rows & 0xffffu);                            // tile_dim1 (tile_dim2=0)
  g1[5] = (int)strideElems;                                 // tensor_dim0_stride lo
  g1[6] = 0;
  g1[7] = 0;
  v4i z4 = {0, 0, 0, 0};
#if __clang_major__ >= 23
  v8i z8 = {0, 0, 0, 0, 0, 0, 0, 0};
  __builtin_amdgcn_tensor_load_to_lds(g0, g1, z4, z4, z8, 0);
#else
  __builtin_amdgcn_tensor_load_to_lds(g0, g1, z4, z4, 0);
#endif
#endif
}

// Load a [rows x rowElems] bf16 panel into LDS (contiguous rows in LDS).
// TDM path: wave 0 issues one descriptor, waits TENSORcnt, then barrier.
__device__ __forceinline__ void load_panel(bf16* dstLds, const bf16* src,
                                           int rows, int rowElems, int srcStride,
                                           int tid, int nthreads) {
#ifdef HAVE_TDM
  if (tid < 32) {
    tdm_load_2d(dstLds, src, (unsigned)rowElems, (unsigned)rows,
                (unsigned)srcStride);
    __builtin_amdgcn_s_wait_tensorcnt((short)0);
  }
#else
  const int cpr = rowElems >> 3;  // U4 chunks per row
  for (int e = tid; e < rows * cpr; e += nthreads) {
    int r = e / cpr, c = e % cpr;
    ((U4*)(dstLds + (size_t)r * rowElems))[c] =
        ((const U4*)(src + (size_t)r * srcStride))[c];
  }
#endif
  __syncthreads();
}

// ---------------- prep: f32 -> bf16 convert ---------------------------------
__global__ void __launch_bounds__(256) k_cvt(const float* __restrict__ src,
                                             bf16* __restrict__ dst, int n) {
  int i = (blockIdx.x * 256 + threadIdx.x) * 4;
  if (i < n) {
    F4 a = *(const F4*)(src + i);
    BF4 o;
#pragma unroll
    for (int j = 0; j < 4; ++j) o.v[j] = (bf16)a.v[j];
    *(BF4*)(dst + i) = o;
  }
}

// ---------------- prep: k_upT[h][r][d] = w_kvb[h*128+d][r] (bf16) -----------
__global__ void __launch_bounds__(256) k_prep(const float* __restrict__ w_kvb,
                                              bf16* __restrict__ k_upT) {
  int e = blockIdx.x * 256 + threadIdx.x;   // H*512*128 = 1048576 exactly
  int h = e >> 16, rem = e & 65535, r = rem >> 7, d = rem & 127;
  k_upT[e] = (bf16)w_kvb[((size_t)(h * 128 + d)) * KV_LORA + r];
}

// ---------------- K1: q_latent = LN(hidden @ w_qa^T) ------------------------
__global__ void __launch_bounds__(256, 2) k_qa_ln(const bf16* __restrict__ hidden,
                                                  const bf16* __restrict__ w_qa,
                                                  const float* __restrict__ g,
                                                  const float* __restrict__ bia,
                                                  bf16* __restrict__ q_latent) {
  extern __shared__ char sm[];
  bf16* As = (bf16*)sm;                       // 16*2048 bf16 = 64KB
  __shared__ float ssum[16], ssq[16], smu[16], srs[16];
  const int tid = threadIdx.x, lane = tid & 31, w = tid >> 5;
  const int hh = lane >> 4, l15 = lane & 15;
  const int row0 = blockIdx.x * 16;

  if (tid < 16) { ssum[tid] = 0.f; ssq[tid] = 0.f; }
  load_panel(As, hidden + (size_t)row0 * D_MODEL, 16, D_MODEL, D_MODEL, tid, 256);

  const bf16* arow = As + l15 * D_MODEL;
  const bf16* bbase[12];
#pragma unroll
  for (int t = 0; t < 12; ++t)
    bbase[t] = w_qa + (size_t)((w * 12 + t) * 16 + l15) * D_MODEL;

  v8f acc[12] = {};
  gemm_panel<12>(arow, bbase, D_MODEL / 32, hh, acc);

#pragma unroll
  for (int t = 0; t < 12; ++t)
#pragma unroll
    for (int j = 0; j < 8; ++j) {
      int r = j + 8 * hh;
      atomicAdd(&ssum[r], acc[t][j]);
      atomicAdd(&ssq[r], acc[t][j] * acc[t][j]);
    }
  __syncthreads();
  if (tid < 16) {
    float mu = ssum[tid] * (1.f / Q_LORA);
    float var = ssq[tid] * (1.f / Q_LORA) - mu * mu;
    smu[tid] = mu; srs[tid] = rsqrtf(var + 1e-5f);
  }
  __syncthreads();
#pragma unroll
  for (int t = 0; t < 12; ++t) {
    int col = (w * 12 + t) * 16 + l15;
    float gv = g[col], bv = bia[col];
#pragma unroll
    for (int j = 0; j < 8; ++j) {
      int r = j + 8 * hh;
      q_latent[(size_t)(row0 + r) * Q_LORA + col] =
          (bf16)((acc[t][j] - smu[r]) * srs[r] * gv + bv);
    }
  }
}

// ------ K2: kv_mixed = hidden @ w_kva^T ; LN -> k_full[:,0:512] & kvT;
//            RoPE -> k_full[:,512:576] ---------------------------------------
__global__ void __launch_bounds__(192, 2) k_kva(const bf16* __restrict__ hidden,
                                                const bf16* __restrict__ w_kva,
                                                const float* __restrict__ g,
                                                const float* __restrict__ bia,
                                                const int* __restrict__ pos_ids,
                                                bf16* __restrict__ k_full,
                                                bf16* __restrict__ kvT) {
  extern __shared__ char sm[];
  bf16* As = (bf16*)sm;                       // 64KB
  __shared__ float ssum[16], ssq[16], smu[16], srs[16];
  __shared__ float ropebuf[16 * 64];
  const int tid = threadIdx.x, lane = tid & 31, w = tid >> 5;
  const int hh = lane >> 4, l15 = lane & 15;
  const int row0 = blockIdx.x * 16;

  if (tid < 16) { ssum[tid] = 0.f; ssq[tid] = 0.f; }
  load_panel(As, hidden + (size_t)row0 * D_MODEL, 16, D_MODEL, D_MODEL, tid, 192);

  const bf16* arow = As + l15 * D_MODEL;
  const bf16* bbase[6];
#pragma unroll
  for (int t = 0; t < 6; ++t)
    bbase[t] = w_kva + (size_t)((w * 6 + t) * 16 + l15) * D_MODEL;

  v8f acc[6] = {};
  gemm_panel<6>(arow, bbase, D_MODEL / 32, hh, acc);

#pragma unroll
  for (int t = 0; t < 6; ++t) {
    int gn = w * 6 + t;
    if (gn < 32) {
#pragma unroll
      for (int j = 0; j < 8; ++j) {
        int r = j + 8 * hh;
        atomicAdd(&ssum[r], acc[t][j]);
        atomicAdd(&ssq[r], acc[t][j] * acc[t][j]);
      }
    } else {
#pragma unroll
      for (int j = 0; j < 8; ++j)
        ropebuf[(j + 8 * hh) * 64 + (gn - 32) * 16 + l15] = acc[t][j];
    }
  }
  __syncthreads();
  if (tid < 16) {
    float mu = ssum[tid] * (1.f / KV_LORA);
    float var = ssq[tid] * (1.f / KV_LORA) - mu * mu;
    smu[tid] = mu; srs[tid] = rsqrtf(var + 1e-5f);
  }
  __syncthreads();
#pragma unroll
  for (int t = 0; t < 6; ++t) {
    int gn = w * 6 + t;
    if (gn >= 32) continue;
    int col = gn * 16 + l15;
    float gv = g[col], bv = bia[col];
#pragma unroll
    for (int j = 0; j < 8; ++j) {
      int r = j + 8 * hh, grow = row0 + r;
      int b = grow >> 11, s = grow & (SEQ - 1);
      bf16 v = (bf16)((acc[t][j] - smu[r]) * srs[r] * gv + bv);
      k_full[(size_t)grow * CD + col] = v;
      kvT[((size_t)(b * KV_LORA + col)) * SEQ + s] = v;
    }
  }
  for (int e = tid; e < 16 * 32; e += 192) {
    int r = e >> 5, j = e & 31;
    int grow = row0 + r, s = grow & (SEQ - 1);
    float ang = (float)pos_ids[s] * __expf(-(float)j * 0.2878231366242557f);
    float sn, cs; __sincosf(ang, &sn, &cs);
    float x1 = ropebuf[r * 64 + j], x2 = ropebuf[r * 64 + 32 + j];
    k_full[(size_t)grow * CD + KV_LORA + j]      = (bf16)(x1 * cs - x2 * sn);
    k_full[(size_t)grow * CD + KV_LORA + 32 + j] = (bf16)(x1 * sn + x2 * cs);
  }
}

// ---------------- K3r: q_rope = RoPE(q_latent @ w_qrope^T) ------------------
__global__ void __launch_bounds__(256, 2) k_qrope(const bf16* __restrict__ qlat,
                                                  const bf16* __restrict__ w_qrope,
                                                  const int* __restrict__ pos_ids,
                                                  bf16* __restrict__ q_full) {
  extern __shared__ char sm[];
  bf16*  As = (bf16*)sm;                      // 16*1536 bf16 = 48KB
  float* Cf = (float*)(sm + 49152);           // 16*1024 f32 = 64KB
  const int tid = threadIdx.x, lane = tid & 31, w = tid >> 5;
  const int hh = lane >> 4, l15 = lane & 15;
  const int row0 = blockIdx.x * 16;

  load_panel(As, qlat + (size_t)row0 * Q_LORA, 16, Q_LORA, Q_LORA, tid, 256);

  const bf16* arow = As + l15 * Q_LORA;
  const bf16* bbase[8];
#pragma unroll
  for (int t = 0; t < 8; ++t)
    bbase[t] = w_qrope + (size_t)((w * 8 + t) * 16 + l15) * Q_LORA;

  v8f acc[8] = {};
  gemm_panel<8>(arow, bbase, Q_LORA / 32, hh, acc);

#pragma unroll
  for (int t = 0; t < 8; ++t)
#pragma unroll
    for (int j = 0; j < 8; ++j)
      Cf[(j + 8 * hh) * 1024 + (w * 8 + t) * 16 + l15] = acc[t][j];
  __syncthreads();
  for (int e = tid; e < 8192; e += 256) {
    int r = e >> 9, rem = e & 511, hd = rem >> 5, j = rem & 31;
    int grow = row0 + r, s = grow & (SEQ - 1);
    float ang = (float)pos_ids[s] * __expf(-(float)j * 0.2878231366242557f);
    float sn, cs; __sincosf(ang, &sn, &cs);
    float x1 = Cf[r * 1024 + hd * 64 + j], x2 = Cf[r * 1024 + hd * 64 + 32 + j];
    size_t base = ((size_t)grow * H + hd) * CD + KV_LORA;
    q_full[base + j]      = (bf16)(x1 * cs - x2 * sn);
    q_full[base + 32 + j] = (bf16)(x1 * sn + x2 * cs);
  }
}

// ---------------- K3n: q_nope = q_latent @ w_qb^T ---------------------------
__global__ void __launch_bounds__(256, 2) k_qnope(const bf16* __restrict__ qlat,
                                                  const bf16* __restrict__ w_qb,
                                                  bf16* __restrict__ q_nope) {
  extern __shared__ char sm[];
  bf16* As = (bf16*)sm;                       // 48KB
  const int tid = threadIdx.x, lane = tid & 31, w = tid >> 5;
  const int hh = lane >> 4, l15 = lane & 15;
  const int row0 = blockIdx.x * 16, col0 = blockIdx.y * 256;

  load_panel(As, qlat + (size_t)row0 * Q_LORA, 16, Q_LORA, Q_LORA, tid, 256);

  const bf16* arow = As + l15 * Q_LORA;
  const bf16* bbase[2];
#pragma unroll
  for (int t = 0; t < 2; ++t)
    bbase[t] = w_qb + (size_t)(col0 + (w * 2 + t) * 16 + l15) * Q_LORA;

  v8f acc[2] = {};
  gemm_panel<2>(arow, bbase, Q_LORA / 32, hh, acc);

#pragma unroll
  for (int t = 0; t < 2; ++t) {
    int col = col0 + (w * 2 + t) * 16 + l15;
#pragma unroll
    for (int j = 0; j < 8; ++j)
      q_nope[(size_t)(row0 + j + 8 * hh) * (H * DN) + col] = (bf16)acc[t][j];
  }
}

// ---------------- K4a: q_abs[b,s,h,:] = q_nope[b,s,h,:] @ k_upT[h] ----------
__global__ void __launch_bounds__(256, 2) k_qabs(const bf16* __restrict__ q_nope,
                                                 const bf16* __restrict__ k_upT,
                                                 bf16* __restrict__ q_full) {
  __shared__ bf16 As[16 * DN];                // 4KB
  const int tid = threadIdx.x, lane = tid & 31, w = tid >> 5;
  const int hh = lane >> 4, l15 = lane & 15;
  const int row0 = blockIdx.x * 16, h = blockIdx.y;

  load_panel(As, q_nope + (size_t)row0 * (H * DN) + h * DN, 16, DN, H * DN, tid, 256);

  const bf16* arow = As + l15 * DN;
  const bf16* bbase[4];
#pragma unroll
  for (int t = 0; t < 4; ++t)
    bbase[t] = k_upT + ((size_t)h * KV_LORA + (w * 4 + t) * 16 + l15) * DN;

  v8f acc[4] = {};
  gemm_panel<4>(arow, bbase, DN / 32, hh, acc);

#pragma unroll
  for (int t = 0; t < 4; ++t) {
    int col = (w * 4 + t) * 16 + l15;
#pragma unroll
    for (int j = 0; j < 8; ++j)
      q_full[((size_t)(row0 + j + 8 * hh) * H + h) * CD + col] = (bf16)acc[t][j];
  }
}

// ---------------- K4: flash attention --------------------------------------
// grid (B*H, SEQ/64), block 256 (8 waves: 4 row-groups x 2 col-groups)
__global__ void __launch_bounds__(256) k_attn(const bf16* __restrict__ q_full,
                                              const bf16* __restrict__ k_full,
                                              const bf16* __restrict__ kvT,
                                              bf16* __restrict__ attn) {
  extern __shared__ char sm[];
  bf16*  Qs   = (bf16*)sm;                             // 64*576*2 = 73728
  float* Sc   = (float*)(sm + 73728);                  // 64*32*4  = 8192
  bf16*  Pp   = (bf16*)(sm + 73728 + 8192);            // 64*32*2  = 4096
  float* mrow = (float*)(sm + 86016);
  float* lrow = (float*)(sm + 86272);
  float* arow = (float*)(sm + 86528);

  const int tid = threadIdx.x, lane = tid & 31, w = tid >> 5;
  const int hh = lane >> 4, l15 = lane & 15;
  const int bh = blockIdx.x, b = bh / H, h = bh % H;
  const int q0 = blockIdx.y * 64;
  const int r4 = w >> 1, cg = w & 1;
  const float scale = 0.0721687836487032f;             // 1/sqrt(192)

  if (tid < 64) { mrow[tid] = -1e30f; lrow[tid] = 0.f; }
  // stage Q tile (64 x 576 bf16) via TDM (row stride in q_full = H*CD)
  load_panel(Qs, q_full + ((size_t)(b * SEQ + q0) * H + h) * CD, 64, CD, H * CD,
             tid, 256);

  v8f o[16] = {};
  const bf16* qrow = Qs + (r4 * 16 + l15) * CD;
  const bf16* vbase = kvT + ((size_t)(b * KV_LORA + cg * 256 + l15)) * SEQ;

  for (int kt = 0; kt < SEQ / 32; ++kt) {
    const int kk = kt * 32;
    // ---- scores tile: 16q x 16k per wave, K=576 (double-buffered K frags)
    v8f sacc = {};
    const bf16* kp = k_full + (size_t)(b * SEQ + kk + cg * 16 + l15) * CD;
    if (kt + 1 < SEQ / 32)
      __builtin_prefetch(k_full + (size_t)(b * SEQ + kk + 32 + cg * 16 + l15) * CD, 0, 0);
    v16bf kb = frag_from_bf16(kp, hh);
#pragma unroll
    for (int kc = 0; kc < CD / 32; ++kc) {
      v16bf knext;
      if (kc + 1 < CD / 32) knext = frag_from_bf16(kp + (kc + 1) * 32, hh);
      v16bf a = frag_from_bf16(qrow + kc * 32, hh);
      sacc = wmma_bf16(a, kb, sacc);
      kb = knext;
    }
#pragma unroll
    for (int j = 0; j < 8; ++j)
      Sc[(r4 * 16 + j + 8 * hh) * 32 + cg * 16 + l15] = sacc[j] * scale;
    __syncthreads();
    // ---- online softmax bookkeeping (one thread per query row)
    if (tid < 64) {
      int r = tid;
      float mloc = -1e30f;
      for (int c = 0; c < 32; ++c) mloc = fmaxf(mloc, Sc[r * 32 + c]);
      float mold = mrow[r], mnew = fmaxf(mold, mloc);
      float al = __expf(mold - mnew);
      float sum = 0.f;
      for (int c = 0; c < 32; ++c) {
        float p = __expf(Sc[r * 32 + c] - mnew);
        Pp[r * 32 + c] = (bf16)p;
        sum += p;
      }
      lrow[r] = lrow[r] * al + sum;
      mrow[r] = mnew;
      arow[r] = al;
    }
    __syncthreads();
    // ---- rescale + P x V accumulate (double-buffered V frags)
    float al8[8];
#pragma unroll
    for (int j = 0; j < 8; ++j) al8[j] = arow[r4 * 16 + j + 8 * hh];
    v16bf pf = frag_from_bf16(Pp + (r4 * 16 + l15) * 32, hh);
    v16bf vf = frag_from_bf16(vbase + kk, hh);
#pragma unroll
    for (int nt = 0; nt < 16; ++nt) {
      v16bf vnext;
      if (nt + 1 < 16)
        vnext = frag_from_bf16(vbase + (size_t)(nt + 1) * 16 * SEQ + kk, hh);
#pragma unroll
      for (int j = 0; j < 8; ++j) o[nt][j] *= al8[j];
      o[nt] = wmma_bf16(pf, vf, o[nt]);
      vf = vnext;
    }
  }
  __syncthreads();
  if (tid < 64) arow[tid] = 1.f / lrow[tid];
  __syncthreads();
  float li8[8];
#pragma unroll
  for (int j = 0; j < 8; ++j) li8[j] = arow[r4 * 16 + j + 8 * hh];
#pragma unroll
  for (int nt = 0; nt < 16; ++nt) {
    int col = cg * 256 + nt * 16 + l15;
#pragma unroll
    for (int j = 0; j < 8; ++j) {
      int r = r4 * 16 + j + 8 * hh;
      attn[((size_t)(b * H + h) * SEQ + q0 + r) * KV_LORA + col] =
          (bf16)(o[nt][j] * li8[j]);
    }
  }
}

// ---------------- K5: attn_v[b,s,h,:] = attn[b,h,s,:] @ v_up[h]^T -----------
__global__ void __launch_bounds__(256, 2) k_attnv(const bf16* __restrict__ attn,
                                                  const bf16* __restrict__ w_kvb,
                                                  bf16* __restrict__ attn_v) {
  __shared__ bf16 As[16 * KV_LORA];           // 16KB
  const int tid = threadIdx.x, lane = tid & 31, w = tid >> 5;
  const int hh = lane >> 4, l15 = lane & 15;
  const int row0 = blockIdx.x * 16, h = blockIdx.y;
  const int b = row0 / SEQ, s0 = row0 % SEQ;

  load_panel(As, attn + ((size_t)(b * H + h) * SEQ + s0) * KV_LORA, 16, KV_LORA,
             KV_LORA, tid, 256);

  const bf16* arow = As + l15 * KV_LORA;
  const bf16* bb = w_kvb + (size_t)(H * DN + h * DV + w * 16 + l15) * KV_LORA;
  const bf16* bbase[1] = { bb };
  v8f acc[1] = {};
  gemm_panel<1>(arow, bbase, KV_LORA / 32, hh, acc);

  int col = h * DV + w * 16 + l15;
#pragma unroll
  for (int j = 0; j < 8; ++j)
    attn_v[(size_t)(row0 + j + 8 * hh) * (H * DV) + col] = (bf16)acc[0][j];
}

// ---------------- K6: out = attn_v @ w_o^T ----------------------------------
__global__ void __launch_bounds__(256, 2) k_out(const bf16* __restrict__ attn_v,
                                                const bf16* __restrict__ w_o,
                                                float* __restrict__ out) {
  extern __shared__ char sm[];
  bf16* As = (bf16*)sm;                       // 64KB
  const int tid = threadIdx.x, lane = tid & 31, w = tid >> 5;
  const int hh = lane >> 4, l15 = lane & 15;
  const int row0 = blockIdx.x * 16, col0 = blockIdx.y * 256;

  load_panel(As, attn_v + (size_t)row0 * (H * DV), 16, H * DV, H * DV, tid, 256);

  const bf16* arow = As + l15 * (H * DV);
  const bf16* bbase[2];
#pragma unroll
  for (int t = 0; t < 2; ++t)
    bbase[t] = w_o + (size_t)(col0 + (w * 2 + t) * 16 + l15) * (H * DV);

  v8f acc[2] = {};
  gemm_panel<2>(arow, bbase, (H * DV) / 32, hh, acc);

#pragma unroll
  for (int t = 0; t < 2; ++t) {
    int col = col0 + (w * 2 + t) * 16 + l15;
#pragma unroll
    for (int j = 0; j < 8; ++j)
      out[(size_t)(row0 + j + 8 * hh) * D_MODEL + col] = acc[t][j];
  }
}

// ---------------- host launch ----------------------------------------------
extern "C" void kernel_launch(void* const* d_in, const int* in_sizes, int n_in,
                              void* d_out, int out_size, void* d_ws, size_t ws_size,
                              hipStream_t stream) {
  const float* hidden   = (const float*)d_in[0];
  const float* w_qa     = (const float*)d_in[1];
  const float* ln_qa_g  = (const float*)d_in[2];
  const float* ln_qa_b  = (const float*)d_in[3];
  const float* w_qb     = (const float*)d_in[4];
  const float* w_qrope  = (const float*)d_in[5];
  const float* w_kva    = (const float*)d_in[6];
  const float* ln_kva_g = (const float*)d_in[7];
  const float* ln_kva_b = (const float*)d_in[8];
  const float* w_kvb    = (const float*)d_in[9];
  const float* w_o      = (const float*)d_in[10];
  const int*   pos      = (const int*)d_in[11];
  float* out = (float*)d_out;

  char* ws = (char*)d_ws;
  size_t off = 0;
  auto alloc = [&](size_t bytes) -> char* {
    char* p = ws + off;
    off += (bytes + 255) & ~(size_t)255;
    return p;
  };
  bf16* q_latent  = (bf16*)alloc((size_t)ROWS * Q_LORA * 2);
  bf16* k_full    = (bf16*)alloc((size_t)ROWS * CD * 2);
  bf16* kvT       = (bf16*)alloc((size_t)BSZ * KV_LORA * SEQ * 2);
  bf16* q_nope    = (bf16*)alloc((size_t)ROWS * H * DN * 2);
  bf16* q_full    = (bf16*)alloc((size_t)ROWS * H * CD * 2);
  bf16* attn      = (bf16*)alloc((size_t)BSZ * H * SEQ * KV_LORA * 2);
  bf16* attn_v    = (bf16*)alloc((size_t)ROWS * H * DV * 2);
  bf16* k_upT     = (bf16*)alloc((size_t)H * KV_LORA * DN * 2);
  bf16* hidden_bf = (bf16*)alloc((size_t)ROWS * D_MODEL * 2);
  bf16* w_qa_bf   = (bf16*)alloc((size_t)Q_LORA * D_MODEL * 2);
  bf16* w_qb_bf   = (bf16*)alloc((size_t)(H * DN) * Q_LORA * 2);
  bf16* w_qr_bf   = (bf16*)alloc((size_t)(H * DR) * Q_LORA * 2);
  bf16* w_kva_bf  = (bf16*)alloc((size_t)CD * D_MODEL * 2);
  bf16* w_kvb_bf  = (bf16*)alloc((size_t)(H * (DN + DV)) * KV_LORA * 2);
  bf16* w_o_bf    = (bf16*)alloc((size_t)D_MODEL * (H * DV) * 2);

  // --- one-time f32 -> bf16 conversions (hoists all cvts out of GEMM loops)
  k_cvt<<<(ROWS * D_MODEL) / 1024, 256, 0, stream>>>(hidden, hidden_bf, ROWS * D_MODEL);
  k_cvt<<<(Q_LORA * D_MODEL) / 1024, 256, 0, stream>>>(w_qa, w_qa_bf, Q_LORA * D_MODEL);
  k_cvt<<<(H * DN * Q_LORA) / 1024, 256, 0, stream>>>(w_qb, w_qb_bf, H * DN * Q_LORA);
  k_cvt<<<(H * DR * Q_LORA) / 1024, 256, 0, stream>>>(w_qrope, w_qr_bf, H * DR * Q_LORA);
  k_cvt<<<(CD * D_MODEL) / 1024, 256, 0, stream>>>(w_kva, w_kva_bf, CD * D_MODEL);
  k_cvt<<<(H * (DN + DV) * KV_LORA) / 1024, 256, 0, stream>>>(w_kvb, w_kvb_bf,
                                                              H * (DN + DV) * KV_LORA);
  k_cvt<<<(D_MODEL * H * DV) / 1024, 256, 0, stream>>>(w_o, w_o_bf, D_MODEL * H * DV);
  k_prep<<<4096, 256, 0, stream>>>(w_kvb, k_upT);

  // --- pipeline
  k_qa_ln<<<ROWS / 16, 256, 16 * D_MODEL * 2, stream>>>(hidden_bf, w_qa_bf, ln_qa_g,
                                                        ln_qa_b, q_latent);
  k_kva  <<<ROWS / 16, 192, 16 * D_MODEL * 2, stream>>>(hidden_bf, w_kva_bf, ln_kva_g,
                                                        ln_kva_b, pos, k_full, kvT);
  k_qrope<<<ROWS / 16, 256, 16 * Q_LORA * 2 + 16 * 1024 * 4, stream>>>(q_latent, w_qr_bf,
                                                                       pos, q_full);
  k_qnope<<<dim3(ROWS / 16, 8), 256, 16 * Q_LORA * 2, stream>>>(q_latent, w_qb_bf,
                                                                q_nope);
  k_qabs <<<dim3(ROWS / 16, H), 256, 0, stream>>>(q_nope, k_upT, q_full);
  k_attn <<<dim3(BSZ * H, SEQ / 64), 256, 86784, stream>>>(q_full, k_full, kvT, attn);
  k_attnv<<<dim3(ROWS / 16, H), 256, 0, stream>>>(attn, w_kvb_bf, attn_v);
  k_out  <<<dim3(ROWS / 16, 8), 256, 16 * H * DV * 2, stream>>>(attn_v, w_o_bf, out);
}
